// Decoder_14233521619067
// MI455X (gfx1250) — matrix operand order repaired and checked by reading, not compiled, not run on hardware
//
#include <hip/hip_runtime.h>
#include <hip/hip_bf16.h>
#include <math.h>

// ---------------------------------------------------------------------------
// Tacotron2-style decoder for MI455X (gfx1250).
// - All big GEMMs via V_WMMA_F32_16X16X32_BF16 (bf16 operands, f32 accum).
// - Weights converted once to bf16 (fit in 192MB L2; halves per-step traffic).
// - Single persistent kernel runs all 500 serial timesteps with a
//   device-wide sense-reversing atomic barrier between dependent phases.
// ---------------------------------------------------------------------------

typedef unsigned short u16;
typedef unsigned int   u32;

typedef __attribute__((ext_vector_type(16))) __bf16 v16bf;
typedef __attribute__((ext_vector_type(8)))  float  v8f;

union OpPack { uint4 q[2]; v16bf v; };   // 32 bytes = 16 bf16 per lane

// ----- problem constants ----------------------------------------------------
constexpr int B = 32, T_ENC = 256, T_DEC = 500;
constexpr int N_MEL = 80, ENC_DIM = 512, ATT_DIM = 128;
constexpr int N_FILT = 32, KSIZE = 31;
constexpr int ATT_RNN = 1024, DEC_RNN = 1024, PRENET = 256;
constexpr int KA = PRENET + ENC_DIM + ATT_RNN;   // 1792 (att-LSTM fused K)
constexpr int KD = ATT_RNN + ENC_DIM + DEC_RNN;  // 2560 (dec-LSTM fused K)
constexpr int GA = 4 * ATT_RNN;                  // 4096 gate width
constexpr int KP1 = 96;                          // prenet-1 K padded 80->96
constexpr int KDHA = ATT_RNN + ENC_DIM;          // 1536

constexpr int NBLK = 64;   // persistent grid: 64 blocks x 256 thr = 512 waves

// ----- output layout (mels | gates | aligns), all f32, [B,T,*] --------------
constexpr size_t MEL_OFF   = 0;
constexpr size_t GATE_OFF  = (size_t)B * T_DEC * N_MEL;            // 1,280,000
constexpr size_t ALIGN_OFF = GATE_OFF + (size_t)B * T_DEC;         // 1,296,000

// ----- workspace layout (bytes) ---------------------------------------------
constexpr size_t al256(size_t x) { return (x + 255) & ~(size_t)255; }
constexpr size_t OFF_WCAT_A = 0;                                        // bf16 [4096,1792]
constexpr size_t OFF_WCAT_D = al256(OFF_WCAT_A + (size_t)GA * KA * 2);  // bf16 [4096,2560]
constexpr size_t OFF_WP1    = al256(OFF_WCAT_D + (size_t)GA * KD * 2);  // bf16 [256,96]
constexpr size_t OFF_WP2    = al256(OFF_WP1   + (size_t)PRENET * KP1 * 2);
constexpr size_t OFF_WQ     = al256(OFF_WP2   + (size_t)PRENET * PRENET * 2);
constexpr size_t OFF_WPROJ  = al256(OFF_WQ    + (size_t)ATT_DIM * ATT_RNN * 2);
constexpr size_t OFF_WMEM   = al256(OFF_WPROJ + (size_t)N_MEL * KDHA * 2);
constexpr size_t OFF_ENCBF  = al256(OFF_WMEM  + (size_t)ATT_DIM * ENC_DIM * 2);
constexpr size_t OFF_PM     = al256(OFF_ENCBF + (size_t)B * T_ENC * ENC_DIM * 2);
constexpr size_t OFF_BCATA  = al256(OFF_PM    + (size_t)B * T_ENC * ATT_DIM * 4);
constexpr size_t OFF_BCATD  = al256(OFF_BCATA + (size_t)GA * 4);
constexpr size_t OFF_WSUM   = al256(OFF_BCATD + (size_t)GA * 4);
constexpr size_t OFF_XP1    = al256(OFF_WSUM  + (size_t)N_FILT * KSIZE * 4);
constexpr size_t OFF_A1     = al256(OFF_XP1   + (size_t)B * KP1 * 2);
constexpr size_t OFF_XCATA  = al256(OFF_A1    + (size_t)B * PRENET * 2);
constexpr size_t OFF_XCATD  = al256(OFF_XCATA + (size_t)B * KA * 2);
constexpr size_t OFF_DHA    = al256(OFF_XCATD + (size_t)B * KD * 2);
constexpr size_t OFF_GATESA = al256(OFF_DHA   + (size_t)B * KDHA * 2);
constexpr size_t OFF_GATESD = al256(OFF_GATESA + (size_t)B * GA * 4);
constexpr size_t OFF_AC     = al256(OFF_GATESD + (size_t)B * GA * 4);
constexpr size_t OFF_DC     = al256(OFF_AC    + (size_t)B * ATT_RNN * 4);
constexpr size_t OFF_PQ     = al256(OFF_DC    + (size_t)B * DEC_RNN * 4);
constexpr size_t OFF_AW     = al256(OFF_PQ    + (size_t)B * ATT_DIM * 4);
constexpr size_t OFF_E      = al256(OFF_AW    + (size_t)B * T_ENC * 4);
constexpr size_t OFF_BAR    = al256(OFF_E     + (size_t)B * T_ENC * 4);
constexpr size_t WS_NEEDED  = OFF_BAR + 256;   // ~51 MB

// ----- helpers --------------------------------------------------------------
__device__ inline u16 f2bf(float f) {
    u32 u = __float_as_uint(f);
    u32 r = (u + 0x7fffu + ((u >> 16) & 1u)) >> 16;   // RNE
    return (u16)r;
}
__device__ inline float bf2f(u16 h) { return __uint_as_float(((u32)h) << 16); }
__device__ inline float sigm(float x) { return 1.f / (1.f + expf(-x)); }

// deterministic "dropout" mask (jax PRNG not reproducible device-side)
__device__ inline float dropmask(u32 t, u32 b, u32 j, u32 salt) {
    u32 x = t * 73856093u ^ b * 19349663u ^ j * 83492791u ^ (salt + 1u) * 2654435761u;
    x ^= x >> 16; x *= 0x7feb352du; x ^= x >> 15; x *= 0x846ca68bu; x ^= x >> 16;
    return (x & 1u) ? 2.0f : 0.0f;
}

// One 16x16 D tile of  X[32,K] @ W[N,K]^T  (bf16 in, f32 accum).
// A lane layout (V_WMMA_*_BF16): lane = (half,m); elems 0..7 -> K = k+half*8..+7,
// elems 8..15 -> K = k+16+half*8..+7.  B: lane = (half,n); same K mapping as A.
// Both are K-contiguous runs => two b128 loads per operand.
__device__ inline v8f gemm_tile(const u16* __restrict__ A, int lda,
                                const u16* __restrict__ W, int ldw,
                                int m0, int n0, int K, int lane) {
    const int half = lane >> 4, l15 = lane & 15;
    const u16* arow = A + (size_t)(m0 + l15) * lda + half * 8;
    const u16* wrow = W + (size_t)(n0 + l15) * ldw + half * 8;
    v8f acc = {0.f, 0.f, 0.f, 0.f, 0.f, 0.f, 0.f, 0.f};
    for (int k = 0; k < K; k += 32) {
        OpPack a, b;
        a.q[0] = *(const uint4*)(arow + k);
        a.q[1] = *(const uint4*)(arow + k + 16);
        b.q[0] = *(const uint4*)(wrow + k);
        b.q[1] = *(const uint4*)(wrow + k + 16);
        acc = __builtin_amdgcn_wmma_f32_16x16x32_bf16(
            false, a.v, false, b.v, (short)0, acc, false, false);
    }
    return acc;
}

// device-wide sense-reversing barrier (grid is co-resident: 64 blocks)
__device__ inline void gbar(int* bar, int* bsense) {
    __syncthreads();
    if (threadIdx.x == 0) {
        __threadfence();
        int s = (*bsense) ^ 1;
        *bsense = s;
        if (atomicAdd(&bar[0], 1) == NBLK - 1) {
            atomicExch(&bar[0], 0);
            __threadfence();
            atomicExch(&bar[1], s);
        } else {
            while (atomicAdd(&bar[1], 0) != s) __builtin_amdgcn_s_sleep(1);
        }
        __threadfence();
    }
    __syncthreads();
}

// ===========================================================================
// prep 1: bf16 weight conversion / fusion / zero-init (graph-capture safe)
// ===========================================================================
__global__ void prep_weights(const float* __restrict__ enc,
                             const float* __restrict__ Wp1, const float* __restrict__ Wp2,
                             const float* __restrict__ Wiha, const float* __restrict__ biha,
                             const float* __restrict__ Whha, const float* __restrict__ bhha,
                             const float* __restrict__ Wq,  const float* __restrict__ Wmem,
                             const float* __restrict__ Wlc,
                             const float* __restrict__ Wihd, const float* __restrict__ bihd,
                             const float* __restrict__ Whhd, const float* __restrict__ bhhd,
                             const float* __restrict__ Wproj, char* __restrict__ ws) {
    size_t tid = (size_t)blockIdx.x * blockDim.x + threadIdx.x;
    size_t stp = (size_t)gridDim.x * blockDim.x;
    u16* wcat_a = (u16*)(ws + OFF_WCAT_A);
    u16* wcat_d = (u16*)(ws + OFF_WCAT_D);
    u16* wp1 = (u16*)(ws + OFF_WP1);    u16* wp2 = (u16*)(ws + OFF_WP2);
    u16* wq = (u16*)(ws + OFF_WQ);      u16* wproj = (u16*)(ws + OFF_WPROJ);
    u16* wmem = (u16*)(ws + OFF_WMEM);  u16* encbf = (u16*)(ws + OFF_ENCBF);
    float* bcat_a = (float*)(ws + OFF_BCATA); float* bcat_d = (float*)(ws + OFF_BCATD);
    float* wsum = (float*)(ws + OFF_WSUM);
    u16* xcat_a = (u16*)(ws + OFF_XCATA); u16* xcat_d = (u16*)(ws + OFF_XCATD);
    u16* dha = (u16*)(ws + OFF_DHA);
    float* ac = (float*)(ws + OFF_AC); float* dc = (float*)(ws + OFF_DC);
    float* aw = (float*)(ws + OFF_AW); int* bar = (int*)(ws + OFF_BAR);

    // fused [W_ih | W_hh] gate weights, bf16 row-major [N,K]
    for (size_t i = tid; i < (size_t)GA * KA; i += stp) {
        size_t n = i / KA, c = i % KA;
        float v = (c < 768) ? Wiha[n * 768 + c] : Whha[n * 1024 + (c - 768)];
        wcat_a[i] = f2bf(v);
    }
    for (size_t i = tid; i < (size_t)GA * KD; i += stp) {
        size_t n = i / KD, c = i % KD;
        float v = (c < 1536) ? Wihd[n * 1536 + c] : Whhd[n * 1024 + (c - 1536)];
        wcat_d[i] = f2bf(v);
    }
    for (size_t i = tid; i < (size_t)PRENET * KP1; i += stp) {
        size_t n = i / KP1, c = i % KP1;
        wp1[i] = f2bf(c < (size_t)N_MEL ? Wp1[n * N_MEL + c] : 0.f);
    }
    for (size_t i = tid; i < (size_t)PRENET * PRENET; i += stp) wp2[i] = f2bf(Wp2[i]);
    for (size_t i = tid; i < (size_t)ATT_DIM * ATT_RNN; i += stp) wq[i] = f2bf(Wq[i]);
    for (size_t i = tid; i < (size_t)N_MEL * KDHA; i += stp) wproj[i] = f2bf(Wproj[i]);
    for (size_t i = tid; i < (size_t)ATT_DIM * ENC_DIM; i += stp) wmem[i] = f2bf(Wmem[i]);
    for (size_t i = tid; i < (size_t)B * T_ENC * ENC_DIM; i += stp) encbf[i] = f2bf(enc[i]);
    for (size_t i = tid; i < (size_t)GA; i += stp) {
        bcat_a[i] = biha[i] + bhha[i];
        bcat_d[i] = bihd[i] + bhhd[i];
    }
    // conv over [aw;aw] stacked channels == (ch0+ch1) filter pre-sum
    for (size_t i = tid; i < (size_t)N_FILT * KSIZE; i += stp) {
        size_t f = i / KSIZE, k = i % KSIZE;
        wsum[i] = Wlc[f * 2 * KSIZE + k] + Wlc[f * 2 * KSIZE + KSIZE + k];
    }
    // zero recurrent state / scratch / barrier
    for (size_t i = tid; i < (size_t)B * KA; i += stp) xcat_a[i] = 0;
    for (size_t i = tid; i < (size_t)B * KD; i += stp) xcat_d[i] = 0;
    for (size_t i = tid; i < (size_t)B * KDHA; i += stp) dha[i] = 0;
    for (size_t i = tid; i < (size_t)B * ATT_RNN; i += stp) { ac[i] = 0.f; dc[i] = 0.f; }
    for (size_t i = tid; i < (size_t)B * T_ENC; i += stp) aw[i] = 0.f;
    for (size_t i = tid; i < 64; i += stp) bar[i] = 0;
}

// ===========================================================================
// prep 2: processed_memory[8192,128] = enc_bf16[8192,512] @ W_mem^T (WMMA)
// ===========================================================================
__global__ void prep_pm(char* __restrict__ ws) {
    const u16* encbf = (const u16*)(ws + OFF_ENCBF);
    const u16* wmem  = (const u16*)(ws + OFF_WMEM);
    float* pm = (float*)(ws + OFF_PM);
    int gtid = blockIdx.x * blockDim.x + threadIdx.x;
    int wave = gtid >> 5, lane = gtid & 31, half = lane >> 4, l15 = lane & 15;
    if (wave >= (B * T_ENC / 16) * (ATT_DIM / 16)) return;     // 512 * 8
    int m0 = (wave >> 3) * 16, n0 = (wave & 7) * 16;
    v8f acc = gemm_tile(encbf, ENC_DIM, wmem, ENC_DIM, m0, n0, ENC_DIM, lane);
    #pragma unroll
    for (int r = 0; r < 8; r++)
        pm[(size_t)(m0 + half * 8 + r) * ATT_DIM + n0 + l15] = acc[r];
}

// ===========================================================================
// persistent decoder: all 500 serial steps in one kernel
// ===========================================================================
__global__ __launch_bounds__(256)
void decoder_loop(const float* __restrict__ enc, const float* __restrict__ dec_in,
                  const float* __restrict__ b_p1, const float* __restrict__ b_p2,
                  const float* __restrict__ Wld, const float* __restrict__ vatt,
                  const float* __restrict__ b_proj, const float* __restrict__ W_gate,
                  const float* __restrict__ b_gate,
                  char* __restrict__ ws, float* __restrict__ out) {
    const u16* wcat_a = (const u16*)(ws + OFF_WCAT_A);
    const u16* wcat_d = (const u16*)(ws + OFF_WCAT_D);
    const u16* wp1 = (const u16*)(ws + OFF_WP1);
    const u16* wp2 = (const u16*)(ws + OFF_WP2);
    const u16* wq  = (const u16*)(ws + OFF_WQ);
    const u16* wproj = (const u16*)(ws + OFF_WPROJ);
    const float* pm = (const float*)(ws + OFF_PM);
    const float* bcat_a = (const float*)(ws + OFF_BCATA);
    const float* bcat_d = (const float*)(ws + OFF_BCATD);
    const float* wsum = (const float*)(ws + OFF_WSUM);
    u16* xp1 = (u16*)(ws + OFF_XP1);
    u16* a1  = (u16*)(ws + OFF_A1);
    u16* xcat_a = (u16*)(ws + OFF_XCATA);
    u16* xcat_d = (u16*)(ws + OFF_XCATD);
    u16* dha = (u16*)(ws + OFF_DHA);
    float* gates_a = (float*)(ws + OFF_GATESA);
    float* gates_d = (float*)(ws + OFF_GATESD);
    float* ac = (float*)(ws + OFF_AC);
    float* dc = (float*)(ws + OFF_DC);
    float* pq = (float*)(ws + OFF_PQ);
    float* aw = (float*)(ws + OFF_AW);
    float* e_buf = (float*)(ws + OFF_E);
    int* bar = (int*)(ws + OFF_BAR);

    __shared__ int bsense;
    __shared__ float sred[256];
    __shared__ float saw[256];
    if (threadIdx.x == 0) bsense = 0;
    __syncthreads();

    const int tid = threadIdx.x, wg = blockIdx.x;
    const int gtid = wg * 256 + tid;
    const int gwave = gtid >> 5, lane = tid & 31;
    const int half = (lane >> 4), l15 = lane & 15;
    const int wv = tid >> 5;

    for (int t = 0; t < T_DEC; t++) {
        // ---- P0/P1: prenet (wg0 only, internal syncs) -----------------------
        if (wg == 0) {
            for (int i = tid; i < B * KP1; i += 256) {      // pack x_t -> bf16
                int b = i / KP1, j = i % KP1;
                float v = (j < N_MEL) ? dec_in[((size_t)b * N_MEL + j) * T_DEC + t] : 0.f;
                xp1[i] = f2bf(v);
            }
            __syncthreads();
            for (int q = 0; q < 4; q++) {                   // fc1: 32 tiles
                int tt = wv * 4 + q;
                int m0 = (tt & 1) * 16, n0 = (tt >> 1) * 16;
                v8f acc = gemm_tile(xp1, KP1, wp1, KP1, m0, n0, KP1, lane);
                int n = n0 + l15; float bb = b_p1[n];
                #pragma unroll
                for (int r = 0; r < 8; r++) {
                    int bm = m0 + half * 8 + r;
                    float v = fmaxf(acc[r] + bb, 0.f) * dropmask(t, bm, n, 0u);
                    a1[bm * PRENET + n] = f2bf(v);
                }
            }
            __syncthreads();
            for (int q = 0; q < 4; q++) {                   // fc2: 32 tiles
                int tt = wv * 4 + q;
                int m0 = (tt & 1) * 16, n0 = (tt >> 1) * 16;
                v8f acc = gemm_tile(a1, PRENET, wp2, PRENET, m0, n0, PRENET, lane);
                int n = n0 + l15; float bb = b_p2[n];
                #pragma unroll
                for (int r = 0; r < 8; r++) {
                    int bm = m0 + half * 8 + r;
                    float v = fmaxf(acc[r] + bb, 0.f) * dropmask(t, bm, n, 1u);
                    xcat_a[bm * KA + n] = f2bf(v);
                }
            }
        }
        gbar(bar, &bsense);                                  // B1

        // ---- P2: att-LSTM gates [32,4096] = xcat_a @ Wcat_a^T (512 tiles) ---
        {
            int m0 = (gwave & 1) * 16, n0 = (gwave >> 1) * 16;
            v8f acc = gemm_tile(xcat_a, KA, wcat_a, KA, m0, n0, KA, lane);
            int n = n0 + l15; float bb = bcat_a[n];
            #pragma unroll
            for (int r = 0; r < 8; r++)
                gates_a[(m0 + half * 8 + r) * GA + n] = acc[r] + bb;
        }
        gbar(bar, &bsense);                                  // B2

        // ---- P3: att-LSTM pointwise -> ah (bf16), ac --------------------------
        for (int i = gtid; i < B * ATT_RNN; i += NBLK * 256) {
            int b = i >> 10, j = i & 1023;
            const float* g = gates_a + b * GA;
            float gi = g[j], gf = g[j + 1024], gg = g[j + 2048], go = g[j + 3072];
            float c = sigm(gf) * ac[i] + sigm(gi) * tanhf(gg);
            ac[i] = c;
            u16 hb = f2bf(sigm(go) * tanhf(c));
            xcat_d[b * KD + j] = hb;                 // dec-LSTM input (this step)
            xcat_a[b * KA + PRENET + ENC_DIM + j] = hb;  // att-LSTM h (next step)
        }
        gbar(bar, &bsense);                                  // B3

        // ---- P4a: query proj pq[32,128] = ah @ W_q^T (wg0, 16 tiles) --------
        if (wg == 0) {
            for (int q = 0; q < 2; q++) {
                int tt = wv * 2 + q;
                int m0 = (tt & 1) * 16, n0 = (tt >> 1) * 16;
                v8f acc = gemm_tile(xcat_d, KD, wq, ATT_RNN, m0, n0, ATT_RNN, lane);
                int n = n0 + l15;
                #pragma unroll
                for (int r = 0; r < 8; r++)
                    pq[(m0 + half * 8 + r) * ATT_DIM + n] = acc[r];
            }
        }
        gbar(bar, &bsense);                                  // B4

        // ---- P4b: location conv + energies (VALU; 8192 threads) -------------
        if (gtid < B * T_ENC) {
            int b = gtid >> 8, tp = gtid & 255;
            float locf[N_FILT];
            #pragma unroll
            for (int f = 0; f < N_FILT; f++) locf[f] = 0.f;
            for (int k = 0; k < KSIZE; k++) {
                int src = tp + k - (KSIZE - 1) / 2;
                if (src >= 0 && src < T_ENC) {
                    float av = aw[b * T_ENC + src];
                    #pragma unroll
                    for (int f = 0; f < N_FILT; f++) locf[f] += wsum[f * KSIZE + k] * av;
                }
            }
            const float* pqb = pq + b * ATT_DIM;
            const float* pmb = pm + ((size_t)b * T_ENC + tp) * ATT_DIM;
            float acc = 0.f;
            for (int d = 0; d < ATT_DIM; d++) {
                float ld = 0.f;
                #pragma unroll
                for (int f = 0; f < N_FILT; f++) ld += Wld[d * N_FILT + f] * locf[f];
                acc += vatt[d] * tanhf(pqb[d] + pmb[d] + ld);
            }
            e_buf[gtid] = acc;
        }
        gbar(bar, &bsense);                                  // B5

        // ---- P4c/d: softmax + context (one wg per batch row) -----------------
        if (wg < B) {
            int b = wg;
            float ev = e_buf[b * T_ENC + tid];
            sred[tid] = ev; __syncthreads();
            for (int s = 128; s > 0; s >>= 1) {
                if (tid < s) sred[tid] = fmaxf(sred[tid], sred[tid + s]);
                __syncthreads();
            }
            float mx = sred[0]; __syncthreads();
            float ex = expf(ev - mx);
            sred[tid] = ex; __syncthreads();
            for (int s = 128; s > 0; s >>= 1) {
                if (tid < s) sred[tid] += sred[tid + s];
                __syncthreads();
            }
            float awv = ex / sred[0];
            aw[b * T_ENC + tid] = awv;
            out[ALIGN_OFF + ((size_t)b * T_DEC + t) * T_ENC + tid] = awv;
            saw[tid] = awv;
            __syncthreads();
            for (int ec = tid; ec < ENC_DIM; ec += 256) {    // context = aw @ enc
                float c = 0.f;
                const float* eb = enc + (size_t)b * T_ENC * ENC_DIM + ec;
                for (int tt = 0; tt < T_ENC; tt++) c += saw[tt] * eb[(size_t)tt * ENC_DIM];
                u16 cb = f2bf(c);
                xcat_d[b * KD + ATT_RNN + ec] = cb;          // dec-LSTM input
                xcat_a[b * KA + PRENET + ec] = cb;           // att-LSTM (next step)
                dha[b * KDHA + DEC_RNN + ec] = cb;           // projection input
            }
        }
        gbar(bar, &bsense);                                  // B6

        // ---- P5: dec-LSTM gates [32,4096] = xcat_d @ Wcat_d^T (512 tiles) ---
        {
            int m0 = (gwave & 1) * 16, n0 = (gwave >> 1) * 16;
            v8f acc = gemm_tile(xcat_d, KD, wcat_d, KD, m0, n0, KD, lane);
            int n = n0 + l15; float bb = bcat_d[n];
            #pragma unroll
            for (int r = 0; r < 8; r++)
                gates_d[(m0 + half * 8 + r) * GA + n] = acc[r] + bb;
        }
        gbar(bar, &bsense);                                  // B7

        // ---- P6: dec-LSTM pointwise -> dh (bf16), dc -------------------------
        for (int i = gtid; i < B * DEC_RNN; i += NBLK * 256) {
            int b = i >> 10, j = i & 1023;
            const float* g = gates_d + b * GA;
            float gi = g[j], gf = g[j + 1024], gg = g[j + 2048], go = g[j + 3072];
            float c = sigm(gf) * dc[i] + sigm(gi) * tanhf(gg);
            dc[i] = c;
            u16 hb = f2bf(sigm(go) * tanhf(c));
            dha[b * KDHA + j] = hb;                          // projection input
            xcat_d[b * KD + ATT_RNN + ENC_DIM + j] = hb;     // dec h (next step)
        }
        gbar(bar, &bsense);                                  // B8

        // ---- P7: mel projection (10 WMMA tiles) + stop gate ------------------
        if (gwave < (B / 16) * (N_MEL / 16)) {               // 10 tiles
            int m0 = (gwave & 1) * 16, n0 = (gwave >> 1) * 16;
            v8f acc = gemm_tile(dha, KDHA, wproj, KDHA, m0, n0, KDHA, lane);
            int n = n0 + l15; float bb = b_proj[n];
            #pragma unroll
            for (int r = 0; r < 8; r++) {
                int brow = m0 + half * 8 + r;
                out[MEL_OFF + ((size_t)brow * T_DEC + t) * N_MEL + n] = acc[r] + bb;
            }
        }
        if (wg == 2 && tid < B) {                            // gate: 32 dots
            float acc = b_gate[0];
            const u16* dr = dha + tid * KDHA;
            for (int k = 0; k < KDHA; k++) acc += bf2f(dr[k]) * W_gate[k];
            out[GATE_OFF + (size_t)tid * T_DEC + t] = acc;
        }
        gbar(bar, &bsense);                                  // B9 -> next t
    }
}

// ===========================================================================
extern "C" void kernel_launch(void* const* d_in, const int* in_sizes, int n_in,
                              void* d_out, int out_size, void* d_ws, size_t ws_size,
                              hipStream_t stream) {
    const float* enc    = (const float*)d_in[0];
    const float* dec_in = (const float*)d_in[1];
    const float* Wp1    = (const float*)d_in[2];
    const float* b_p1   = (const float*)d_in[3];
    const float* Wp2    = (const float*)d_in[4];
    const float* b_p2   = (const float*)d_in[5];
    const float* Wiha   = (const float*)d_in[6];
    const float* biha   = (const float*)d_in[7];
    const float* Whha   = (const float*)d_in[8];
    const float* bhha   = (const float*)d_in[9];
    const float* Wq     = (const float*)d_in[10];
    const float* Wmem   = (const float*)d_in[11];
    const float* vatt   = (const float*)d_in[12];
    const float* Wlc    = (const float*)d_in[13];
    const float* Wld    = (const float*)d_in[14];
    const float* Wihd   = (const float*)d_in[15];
    const float* bihd   = (const float*)d_in[16];
    const float* Whhd   = (const float*)d_in[17];
    const float* bhhd   = (const float*)d_in[18];
    const float* Wproj  = (const float*)d_in[19];
    const float* b_proj = (const float*)d_in[20];
    const float* Wgate  = (const float*)d_in[21];
    const float* b_gate = (const float*)d_in[22];
    char* ws = (char*)d_ws;
    float* out = (float*)d_out;
    (void)n_in; (void)in_sizes; (void)out_size; (void)ws_size; // needs WS_NEEDED (~51MB)

    prep_weights<<<2048, 256, 0, stream>>>(enc, Wp1, Wp2, Wiha, biha, Whha, bhha,
                                           Wq, Wmem, Wlc, Wihd, bihd, Whhd, bhhd,
                                           Wproj, ws);
    prep_pm<<<512, 256, 0, stream>>>(ws);
    decoder_loop<<<NBLK, 256, 0, stream>>>(enc, dec_in, b_p1, b_p2, Wld, vatt,
                                           b_proj, Wgate, b_gate, ws, out);
}
